// CoKEModel_77730318123160
// MI455X (gfx1250) — matrix operand order repaired and checked by reading, not compiled, not run on hardware
//
#include <hip/hip_runtime.h>
#include <hip/hip_bf16.h>

// ---------------------------------------------------------------------------
// Types for CDNA5 WMMA (wave32, V_WMMA_F32_16X16X32_F16)
// ---------------------------------------------------------------------------
typedef __attribute__((ext_vector_type(8)))  _Float16 h8_t;
typedef __attribute__((ext_vector_type(16))) _Float16 h16_t;
typedef __attribute__((ext_vector_type(8)))  float    f8_t;

#define BLK   128   // 4 waves
#define TILE  64
#define KT    32
#define LDSS  40    // f16 elements per LDS row (32 + 8 pad -> 80B stride)

// ---------------------------------------------------------------------------
// gfx1250 async global->LDS helpers (ASYNCcnt-tracked, LDS-direct)
// ---------------------------------------------------------------------------
__device__ __forceinline__ void async_cp32B(const _Float16* g, _Float16* l) {
  // copy 32 bytes (16 f16): two b128 async transfers; inst offset applies to
  // both the LDS and memory addresses, so one address pair serves both.
  unsigned lds = (unsigned)(unsigned long long)l;
  asm volatile("global_load_async_to_lds_b128 %0, %1, off"
               :: "v"(lds), "v"(g) : "memory");
  asm volatile("global_load_async_to_lds_b128 %0, %1, off offset:16"
               :: "v"(lds), "v"(g) : "memory");
}

__device__ __forceinline__ void wait_async0() {
#if __has_builtin(__builtin_amdgcn_s_wait_asynccnt)
  __builtin_amdgcn_s_wait_asynccnt(0);
#else
  asm volatile("s_wait_asynccnt 0x0" ::: "memory");
#endif
}

// ---------------------------------------------------------------------------
// Generic batched WMMA GEMM:  C = act(alpha * A x B^T + bias)
//   A logical (M,K):  TA=0 -> a[m*lda+k],  TA=1 -> a[k*lda+m]
//   B logical (N,K):  TB=0 -> b[n*ldb+k],  TB=1 -> b[k*ldb+n]   (TB=1 == "A@B")
//   batch z: offset = (z/bH)*s?1 + (z%bH)*s?2   (elements)
// ---------------------------------------------------------------------------
struct GemmP {
  const _Float16* A;
  const _Float16* Bw;
  const float* bias;
  float*    Cf;
  _Float16* Ch;
  int M, N, K, lda, ldb, ldc, act, bH;
  float alpha;
  long long sa1, sa2, sb1, sb2, sc1, sc2;
};

__device__ __forceinline__ void wmma_tile(const _Float16* As, const _Float16* Bs,
                                          f8_t acc[2][2], int wm, int wn,
                                          int m16, int hlf) {
  // fragments per ISA 7.12.2 (16-bit A 16x32; B gathered in same K order as A)
  h16_t af[2], bf[2];
  for (int s = 0; s < 2; ++s) {
    const _Float16* ap = &As[(wm + s * 16 + m16) * LDSS];
    h8_t alo = *(const h8_t*)(ap + hlf * 8);
    h8_t ahi = *(const h8_t*)(ap + 16 + hlf * 8);
    for (int e = 0; e < 8; ++e) { af[s][e] = alo[e]; af[s][8 + e] = ahi[e]; }
    const _Float16* bp = &Bs[(wn + s * 16 + m16) * LDSS];
    h8_t blo = *(const h8_t*)(bp + hlf * 8);
    h8_t bhi = *(const h8_t*)(bp + 16 + hlf * 8);
    for (int e = 0; e < 8; ++e) { bf[s][e] = blo[e]; bf[s][8 + e] = bhi[e]; }
  }
  for (int i = 0; i < 2; ++i)
    for (int j = 0; j < 2; ++j)
      acc[i][j] = __builtin_amdgcn_wmma_f32_16x16x32_f16(
          false, af[i], false, bf[j], (short)0, acc[i][j], false, false);
}

template <int TA, int TB>
__global__ __launch_bounds__(BLK) void gemm_wmma_t(GemmP p) {
  __shared__ alignas(16) _Float16 AsBuf[2][TILE * LDSS];
  __shared__ alignas(16) _Float16 BsBuf[2][TILE * LDSS];

  const int bz = blockIdx.z;
  const int b1 = bz / p.bH, b2 = bz % p.bH;
  const _Float16* A  = p.A  + (long long)b1 * p.sa1 + (long long)b2 * p.sa2;
  const _Float16* Bw = p.Bw + (long long)b1 * p.sb1 + (long long)b2 * p.sb2;
  const long long coff = (long long)b1 * p.sc1 + (long long)b2 * p.sc2;
  float*    Cf = p.Cf ? p.Cf + coff : nullptr;
  _Float16* Ch = p.Ch ? p.Ch + coff : nullptr;

  const int row0 = blockIdx.y * TILE;
  const int col0 = blockIdx.x * TILE;
  const int tid  = threadIdx.x;
  const int wave = tid >> 5;
  const int lane = tid & 31;
  const int wm   = (wave >> 1) * 32;
  const int wn   = (wave & 1) * 32;
  const int m16  = lane & 15;
  const int hlf  = lane >> 4;

  f8_t acc[2][2];
  for (int i = 0; i < 2; ++i)
    for (int j = 0; j < 2; ++j)
      for (int e = 0; e < 8; ++e) acc[i][j][e] = 0.f;

  bool fast = false;
  if (TA == 0 && TB == 0)
    fast = (p.lda % 8 == 0) && (p.ldb % 8 == 0) &&
           ((((unsigned long long)A) & 15ULL) == 0) &&
           ((((unsigned long long)Bw) & 15ULL) == 0) &&
           (row0 + TILE <= p.M) && (col0 + TILE <= p.N) && (p.K % KT == 0);

  if (TA == 0 && TB == 0 && fast) {
    // ---------- async double-buffered path (global_load_async_to_lds) ------
    const int r  = tid >> 1;        // tile row handled by this thread
    const int hk = tid & 1;         // which 16-element half of the K-tile
    const _Float16* Ap = A  + (long long)(row0 + r) * p.lda + hk * 16;
    const _Float16* Bp = Bw + (long long)(col0 + r) * p.ldb + hk * 16;
    _Float16* la[2] = { &AsBuf[0][r * LDSS + hk * 16], &AsBuf[1][r * LDSS + hk * 16] };
    _Float16* lb[2] = { &BsBuf[0][r * LDSS + hk * 16], &BsBuf[1][r * LDSS + hk * 16] };

    async_cp32B(Ap, la[0]);
    async_cp32B(Bp, lb[0]);
    wait_async0();
    __syncthreads();

    const int nT = p.K / KT;
    for (int t = 0; t < nT; ++t) {
      const int cur = t & 1;
      if (t + 1 < nT) {
        async_cp32B(Ap + (long long)(t + 1) * KT, la[cur ^ 1]);
        async_cp32B(Bp + (long long)(t + 1) * KT, lb[cur ^ 1]);
      }
      wmma_tile(AsBuf[cur], BsBuf[cur], acc, wm, wn, m16, hlf);
      wait_async0();
      __syncthreads();
    }
  } else {
    // ---------- guarded scalar path (edge tiles / transposed / odd strides)
    for (int k0 = 0; k0 < p.K; k0 += KT) {
      for (int idx = tid; idx < TILE * KT; idx += BLK) {
        int r = idx >> 5;
        int c = idx & 31;
        int gk = k0 + c;
        _Float16 va = (_Float16)0.f;
        int gm = row0 + r;
        if (gm < p.M && gk < p.K)
          va = TA ? A[(long long)gk * p.lda + gm] : A[(long long)gm * p.lda + gk];
        AsBuf[0][r * LDSS + c] = va;
        _Float16 vb = (_Float16)0.f;
        int gn = col0 + r;
        if (gn < p.N && gk < p.K)
          vb = TB ? Bw[(long long)gk * p.ldb + gn] : Bw[(long long)gn * p.ldb + gk];
        BsBuf[0][r * LDSS + c] = vb;
      }
      __syncthreads();
      if (!TA && (k0 + KT) < p.K)
        __builtin_prefetch(A + (long long)(row0 + (tid & 63)) * p.lda + (k0 + KT), 0, 1);
      wmma_tile(AsBuf[0], BsBuf[0], acc, wm, wn, m16, hlf);
      __syncthreads();
    }
  }

  // epilogue: C/D layout -> row = 8*hlf + v, col = lane%16
  for (int i = 0; i < 2; ++i)
    for (int j = 0; j < 2; ++j)
      for (int v = 0; v < 8; ++v) {
        int r = row0 + wm + i * 16 + hlf * 8 + v;
        int c = col0 + wn + j * 16 + m16;
        if (r < p.M && c < p.N) {
          float val = acc[i][j][v] * p.alpha;
          if (p.bias) val += p.bias[c];
          if (p.act == 1) val = fmaxf(val, 0.f);
          else if (p.act == 2) val = 0.5f * val * (1.f + erff(val * 0.70710678118f));
          long long o = (long long)r * p.ldc + c;
          if (Cf) Cf[o] = val;
          if (Ch) Ch[o] = (_Float16)val;
        }
      }
}

// ---------------------------------------------------------------------------
// Small helper kernels
// ---------------------------------------------------------------------------
__device__ __forceinline__ float blk_sum(float v, float* red) {
  int t = threadIdx.x, n = blockDim.x;
  red[t] = v; __syncthreads();
  for (int s = n >> 1; s > 0; s >>= 1) { if (t < s) red[t] += red[t + s]; __syncthreads(); }
  float r = red[0]; __syncthreads();
  return r;
}
__device__ __forceinline__ float blk_max(float v, float* red) {
  int t = threadIdx.x, n = blockDim.x;
  red[t] = v; __syncthreads();
  for (int s = n >> 1; s > 0; s >>= 1) { if (t < s) red[t] = fmaxf(red[t], red[t + s]); __syncthreads(); }
  float r = red[0]; __syncthreads();
  return r;
}

__global__ void f32_to_f16_k(const float* s, _Float16* d, long long n) {
  long long i = (long long)blockIdx.x * blockDim.x + threadIdx.x;
  if (i < n) d[i] = (_Float16)s[i];
}

__global__ void gather_embed_k(const int* tok, const float* emb, _Float16* outb,
                               int nrows, int rowbase) {
  int row = blockIdx.x, tid = threadIdx.x;
  _Float16* dst = outb + (long long)row * 1024;
  if (row < nrows) {
    int t = tok[(long long)rowbase + row];
    t = t < 0 ? 0 : (t > 19999 ? 19999 : t);
    const float* srcp = emb + (long long)t * 1024;
    for (int c = tid; c < 1024; c += blockDim.x) dst[c] = (_Float16)srcp[c];
  } else {
    for (int c = tid; c < 1024; c += blockDim.x) dst[c] = (_Float16)0.f;
  }
}

__global__ void maxpool_k(const float* y1, const float* y2, const float* y3,
                          _Float16* pooled, int nseq, int seqbase) {
  int idx = blockIdx.x * blockDim.x + threadIdx.x;
  if (idx >= nseq * 300) return;
  int seq = idx / 300, f = idx % 300;
  const float* y; int valid, oc;
  if (f < 100)      { y = y1; valid = 50; oc = f; }
  else if (f < 200) { y = y2; valid = 49; oc = f - 100; }
  else              { y = y3; valid = 48; oc = f - 200; }
  float m = -1e30f;
  for (int pos = 0; pos < valid; ++pos)
    m = fmaxf(m, y[(long long)(seq * 50 + pos) * 100 + oc]);
  pooled[(long long)(seqbase + seq) * 300 + f] = (_Float16)m;
}

__global__ void ln_free_k(const float* x, float* of, _Float16* oh, int D) {
  __shared__ float red[256];
  int row = blockIdx.x, tid = threadIdx.x;
  long long base = (long long)row * D;
  float s = 0.f;
  for (int c = tid; c < D; c += blockDim.x) s += x[base + c];
  float mu = blk_sum(s, red) / D;
  float vv = 0.f;
  for (int c = tid; c < D; c += blockDim.x) { float d = x[base + c] - mu; vv += d * d; }
  float var = blk_sum(vv, red) / D;
  float rs = rsqrtf(var + 1e-12f);
  for (int c = tid; c < D; c += blockDim.x) {
    float y = (x[base + c] - mu) * rs;
    if (of) of[base + c] = y;
    if (oh) oh[base + c] = (_Float16)y;
  }
}

__global__ void ln_res_k(float* x, const float* delta, const float* g, const float* bb,
                         _Float16* xh, int D) {
  __shared__ float red[256];
  int row = blockIdx.x, tid = threadIdx.x;
  long long base = (long long)row * D;
  float s = 0.f;
  for (int c = tid; c < D; c += blockDim.x) s += x[base + c] + delta[base + c];
  float mu = blk_sum(s, red) / D;
  float vv = 0.f;
  for (int c = tid; c < D; c += blockDim.x) {
    float d = x[base + c] + delta[base + c] - mu; vv += d * d;
  }
  float var = blk_sum(vv, red) / D;
  float rs = rsqrtf(var + 1e-5f);
  for (int c = tid; c < D; c += blockDim.x) {
    float y = (x[base + c] + delta[base + c] - mu) * rs * g[c] + bb[c];
    x[base + c] = y; xh[base + c] = (_Float16)y;
  }
}

__global__ void ln_scale_k(const float* x, const float* ps, const float* pb,
                           float* of, _Float16* oh, int S, int D) {
  __shared__ float red[256];
  int row = blockIdx.x, tid = threadIdx.x;
  int b = row / S, pos = row % S;
  long long base  = (long long)row * D;
  long long pbase = ((long long)b * 300 + pos) * D;
  float s = 0.f;
  for (int c = tid; c < D; c += blockDim.x) s += x[base + c];
  float mu = blk_sum(s, red) / D;
  float vv = 0.f;
  for (int c = tid; c < D; c += blockDim.x) { float d = x[base + c] - mu; vv += d * d; }
  float var = blk_sum(vv, red) / D;
  float rs = rsqrtf(var + 1e-12f);
  for (int c = tid; c < D; c += blockDim.x) {
    float y = (x[base + c] - mu) * rs * ps[pbase + c] + pb[pbase + c];
    of[base + c] = y; oh[base + c] = (_Float16)y;
  }
}

__global__ void build_s1_k(const float* s2, float* of, _Float16* oh) {
  int row = blockIdx.x, tid = threadIdx.x;
  int b = row / 101, pos = row % 101;
  long long dbase = (long long)row * 512;
  if (pos == 0) {
    for (int c = tid; c < 512; c += blockDim.x) { of[dbase + c] = 0.f; oh[dbase + c] = (_Float16)0.f; }
  } else {
    const float* srcr = s2 + ((long long)b * 200 + (pos - 1)) * 512;
    for (int c = tid; c < 512; c += blockDim.x) {
      float v = srcr[c]; of[dbase + c] = v; oh[dbase + c] = (_Float16)v;
    }
  }
}

__global__ void attn_softmax_k(const float* sc, const unsigned char* mask,
                               _Float16* pr, int S) {
  __shared__ float red[256];
  int row = blockIdx.x, tid = threadIdx.x;
  long long base = (long long)row * S;
  float mx = -1e30f;
  for (int j = tid; j < S; j += blockDim.x) {
    float v = mask[base + j] ? -1e9f : sc[base + j];
    mx = fmaxf(mx, v);
  }
  mx = blk_max(mx, red);
  float sum = 0.f;
  for (int j = tid; j < S; j += blockDim.x) {
    float v = mask[base + j] ? -1e9f : sc[base + j];
    sum += expf(v - mx);
  }
  sum = blk_sum(sum, red);
  float inv = 1.f / sum;
  for (int j = tid; j < S; j += blockDim.x) {
    float v = mask[base + j] ? -1e9f : sc[base + j];
    pr[base + j] = (_Float16)(expf(v - mx) * inv);
  }
}

__global__ void softmax_rows_k(const float* srcp, long long rstride, float* dst, int D) {
  __shared__ float red[256];
  int row = blockIdx.x, tid = threadIdx.x;
  const float* s = srcp + (long long)row * rstride;
  float mx = -1e30f;
  for (int c = tid; c < D; c += blockDim.x) mx = fmaxf(mx, s[c]);
  mx = blk_max(mx, red);
  float sum = 0.f;
  for (int c = tid; c < D; c += blockDim.x) sum += expf(s[c] - mx);
  sum = blk_sum(sum, red);
  for (int c = tid; c < D; c += blockDim.x) dst[(long long)row * D + c] = expf(s[c] - mx) / sum;
}

__global__ void chunk_mean_softmax_k(const float* srcp, int D, int S, float* outp) {
  __shared__ float buf[512];
  __shared__ float red[256];
  int blk = blockIdx.x, tid = threadIdx.x;
  int b = blk >> 1, i = blk & 1;
  int start = i, end = (i + 1) * 100, cnt = end - start;
  for (int c = tid; c < D; c += blockDim.x) {
    float acc = 0.f;
    const float* base = srcp + ((long long)b * S + start) * D + c;
    for (int r = 0; r < cnt; ++r) acc += base[(long long)r * D];
    buf[c] = acc / cnt;
  }
  __syncthreads();
  float mx = -1e30f;
  for (int c = tid; c < D; c += blockDim.x) mx = fmaxf(mx, buf[c]);
  mx = blk_max(mx, red);
  float sum = 0.f;
  for (int c = tid; c < D; c += blockDim.x) sum += expf(buf[c] - mx);
  sum = blk_sum(sum, red);
  for (int c = tid; c < D; c += blockDim.x)
    outp[((long long)b * 2 + i) * D + c] = expf(buf[c] - mx) / sum;
}

__global__ void loss_sims_k(const float* cmean, const float* amean, float* outp) {
  __shared__ float red[32];
  int b = threadIdx.x;
  const float eps = 1e-6f;
  float kl[4], eu[4];
  for (int i = 0; i < 2; ++i)
    for (int j = 0; j < 2; ++j) {
      const float* c1 = cmean + ((long long)b * 2 + i) * 204;
      const float* c2 = cmean + ((long long)b * 2 + j) * 204;
      float s = 0.f;
      for (int m = 0; m < 204; ++m) s += (c1[m] + eps) * logf((c1[m] + eps) / (c2[m] + eps));
      kl[i * 2 + j] = s * 10.f;
      const float* a1 = amean + ((long long)b * 2 + i) * 512;
      const float* a2 = amean + ((long long)b * 2 + j) * 512;
      float e2 = 0.f;
      for (int m = 0; m < 512; ++m) { float d = a1[m] - a2[m]; e2 += d * d; }
      eu[i * 2 + j] = sqrtf(e2 + 1e-12f);
    }
  float mk = fmaxf(fmaxf(kl[0], kl[1]), fmaxf(kl[2], kl[3]));
  float me = fmaxf(fmaxf(eu[0], eu[1]), fmaxf(eu[2], eu[3]));
  float sk = 0.f, se = 0.f;
  for (int t = 0; t < 4; ++t) { kl[t] = expf(kl[t] - mk); sk += kl[t];
                                eu[t] = expf(eu[t] - me); se += eu[t]; }
  float L = 0.f;
  for (int t = 0; t < 4; ++t) {
    float a = kl[t] / sk + eps, c = eu[t] / se + eps;
    L += a * logf(a / c);
  }
  red[b] = L; __syncthreads();
  for (int s = 16; s > 0; s >>= 1) { if (b < s) red[b] += red[b + s]; __syncthreads(); }
  if (b == 0) outp[0] = (red[0] / 32.f) * 100000.f;
}

__global__ void clsIII_k(const float* p101f, const float* fc3w, const float* fc3b, float* outc) {
  int b = blockIdx.x, n = threadIdx.x;
  if (n >= 204) return;
  const float* x = p101f + (long long)b * 101 * 204;
  float s = fc3b[n];
  for (int m = 0; m < 204; ++m) s += x[m] * fc3w[n * 204 + m];
  outc[b * 204 + n] = s;
}

__global__ void outIII_k(const float* clsIII, const float* lab, const float* SLl, float* o) {
  int b = blockIdx.x, n = threadIdx.x;
  if (n >= 204) return;
  const float* cb = clsIII + b * 204;
  const float* sl = SLl + (long long)b * 204 * 204;
  float s = 0.f;
  for (int m = 0; m < 204; ++m) s += cb[m] * (lab[m * 204 + n] + 0.4f * sl[m * 204 + n]);
  o[b * 204 + n] = s;
}

__global__ void contrastive_k(const float* p101f, const float* user4m, float* outp) {
  __shared__ float ncls[32], nusr[32];
  __shared__ float sc[32][32];
  __shared__ float red[1024];
  int t = threadIdx.x;
  if (t < 32) {
    const float* r = p101f + (long long)t * 101 * 204;
    float s = 0.f; for (int m = 0; m < 204; ++m) s += r[m] * r[m];
    ncls[t] = fmaxf(sqrtf(s), 1e-12f);
  } else if (t < 64) {
    const float* r = user4m + (long long)(t - 32) * 204;
    float s = 0.f; for (int m = 0; m < 204; ++m) s += r[m] * r[m];
    nusr[t - 32] = fmaxf(sqrtf(s), 1e-12f);
  }
  __syncthreads();
  int i = t >> 5, j = t & 31;
  const float* ri = p101f + (long long)i * 101 * 204;
  const float* rj = user4m + (long long)j * 204;
  float d = 0.f;
  for (int m = 0; m < 204; ++m) d += ri[m] * rj[m];
  sc[i][j] = d / (ncls[i] * nusr[j]);
  __syncthreads();
  float v = 0.f;
  if (i != j)
    v = fmaxf(0.02f + sc[i][j] - sc[i][i], 0.f) + fmaxf(0.02f + sc[i][j] - sc[j][j], 0.f);
  red[t] = v; __syncthreads();
  for (int s = 512; s > 0; s >>= 1) { if (t < s) red[t] += red[t + s]; __syncthreads(); }
  if (t == 0) outp[0] = red[0] * 0.01f;
}

// ---------------------------------------------------------------------------
// Host-side helpers
// ---------------------------------------------------------------------------
static inline int ceil_div_i(int a, int b) { return (a + b - 1) / b; }

static void gemm(hipStream_t st, const _Float16* A, const _Float16* Bw, const float* bias,
                 float* Cf, _Float16* Ch, int M, int N, int K,
                 int lda, int ldb, int ldc, int ta, int tb, float alpha, int act,
                 int nb = 1, int bH = 1,
                 long long sa1 = 0, long long sa2 = 0, long long sb1 = 0, long long sb2 = 0,
                 long long sc1 = 0, long long sc2 = 0) {
  GemmP p{A, Bw, bias, Cf, Ch, M, N, K, lda, ldb, ldc, act, bH, alpha,
          sa1, sa2, sb1, sb2, sc1, sc2};
  dim3 g(ceil_div_i(N, TILE), ceil_div_i(M, TILE), nb);
  if (!ta && !tb)      gemm_wmma_t<0, 0><<<g, BLK, 0, st>>>(p);
  else if (!ta && tb)  gemm_wmma_t<0, 1><<<g, BLK, 0, st>>>(p);
  else if (ta && tb)   gemm_wmma_t<1, 1><<<g, BLK, 0, st>>>(p);
  else                 gemm_wmma_t<1, 0><<<g, BLK, 0, st>>>(p);
}

struct EncCtx {
  hipStream_t st;
  const _Float16 *wq16, *wk16, *wv16, *wo16, *fw1_16, *fw2_16;
  const float *bq, *bk, *bv, *bo, *l1g, *l1b, *fb1, *fb2, *l2g, *l2b;
  _Float16 *q16, *k16, *v16, *probs16, *ctx16, *h16;
  float *scoresf, *delta;
};

static void encode(const EncCtx& c, float* xf, _Float16* xh, int S, const unsigned char* mask) {
  const int M = 32 * S;
  for (int l = 0; l < 4; ++l) {
    const _Float16* Wq = c.wq16 + (long long)l * 512 * 512;
    const _Float16* Wk = c.wk16 + (long long)l * 512 * 512;
    const _Float16* Wv = c.wv16 + (long long)l * 512 * 512;
    const _Float16* Wo = c.wo16 + (long long)l * 512 * 512;
    gemm(c.st, xh, Wq, c.bq + l * 512, nullptr, c.q16, M, 512, 512, 512, 512, 512, 0, 0, 1.f, 0);
    gemm(c.st, xh, Wk, c.bk + l * 512, nullptr, c.k16, M, 512, 512, 512, 512, 512, 0, 0, 1.f, 0);
    gemm(c.st, xh, Wv, c.bv + l * 512, nullptr, c.v16, M, 512, 512, 512, 512, 512, 0, 0, 1.f, 0);
    // scores[b,h] = (1/8) * Q Kt : batched over 32*8, NT
    gemm(c.st, c.q16, c.k16, nullptr, c.scoresf, nullptr, S, S, 64, 512, 512, S, 0, 0, 0.125f, 0,
         256, 8, (long long)S * 512, 64, (long long)S * 512, 64,
         (long long)8 * S * S, (long long)S * S);
    attn_softmax_k<<<dim3(32 * 8 * S), 256, 0, c.st>>>(c.scoresf, mask, c.probs16, S);
    // ctx[b,h] = P V : batched, NN
    gemm(c.st, c.probs16, c.v16, nullptr, nullptr, c.ctx16, S, 64, S, S, 512, 512, 0, 1, 1.f, 0,
         256, 8, (long long)8 * S * S, (long long)S * S, (long long)S * 512, 64,
         (long long)S * 512, 64);
    gemm(c.st, c.ctx16, Wo, c.bo + l * 512, c.delta, nullptr, M, 512, 512, 512, 512, 512, 0, 0, 1.f, 0);
    ln_res_k<<<dim3(M), 256, 0, c.st>>>(xf, c.delta, c.l1g + l * 512, c.l1b + l * 512, xh, 512);
    gemm(c.st, xh, c.fw1_16 + (long long)l * 2048 * 512, c.fb1 + l * 2048, nullptr, c.h16,
         M, 2048, 512, 512, 512, 2048, 0, 0, 1.f, 1);
    gemm(c.st, c.h16, c.fw2_16 + (long long)l * 2048 * 512, c.fb2 + l * 512, c.delta, nullptr,
         M, 512, 2048, 2048, 2048, 512, 0, 0, 1.f, 0);
    ln_res_k<<<dim3(M), 256, 0, c.st>>>(xf, c.delta, c.l2g + l * 512, c.l2b + l * 512, xh, 512);
  }
}

// ---------------------------------------------------------------------------
// kernel_launch
// ---------------------------------------------------------------------------
extern "C" void kernel_launch(void* const* d_in, const int* in_sizes, int n_in,
                              void* d_out, int out_size, void* d_ws, size_t ws_size,
                              hipStream_t stream) {
  (void)in_sizes; (void)n_in; (void)out_size;
  const int*   src       = (const int*)d_in[0];
  const float* user_info = (const float*)d_in[1];
  const float* label_sim = (const float*)d_in[2];
  const float* emb       = (const float*)d_in[3];
  const float* cw1 = (const float*)d_in[4];  const float* cb1 = (const float*)d_in[5];
  const float* cw2 = (const float*)d_in[6];  const float* cb2 = (const float*)d_in[7];
  const float* cw3 = (const float*)d_in[8];  const float* cb3 = (const float*)d_in[9];
  const float* tfw = (const float*)d_in[10]; const float* tfb = (const float*)d_in[11];
  const float* wq  = (const float*)d_in[12]; const float* bq  = (const float*)d_in[13];
  const float* wk  = (const float*)d_in[14]; const float* bk  = (const float*)d_in[15];
  const float* wv  = (const float*)d_in[16]; const float* bv  = (const float*)d_in[17];
  const float* wo  = (const float*)d_in[18]; const float* bo  = (const float*)d_in[19];
  const float* l1g = (const float*)d_in[20]; const float* l1b = (const float*)d_in[21];
  const float* fw1 = (const float*)d_in[22]; const float* fb1 = (const float*)d_in[23];
  const float* fw2 = (const float*)d_in[24]; const float* fb2 = (const float*)d_in[25];
  const float* l2g = (const float*)d_in[26]; const float* l2b = (const float*)d_in[27];
  const float* fc1w = (const float*)d_in[28]; const float* fc1b = (const float*)d_in[29];
  const float* fc2w = (const float*)d_in[30]; const float* fc2b = (const float*)d_in[31];
  const float* fc3w = (const float*)d_in[32]; const float* fc3b = (const float*)d_in[33];
  const float* fc4w = (const float*)d_in[34]; const float* fc4b = (const float*)d_in[35];
  const float* pscale = (const float*)d_in[36]; const float* pbias = (const float*)d_in[37];
  const unsigned char* mask101 = (const unsigned char*)d_in[38];
  const unsigned char* mask200 = (const unsigned char*)d_in[39];
  float* outp = (float*)d_out;

  // bump allocator over d_ws (reset every call -> deterministic)
  size_t off = 0;
  auto alloc = [&](size_t bytes) -> void* {
    off = (off + 255) & ~(size_t)255;
    void* p = (char*)d_ws + off;
    off += bytes;
    return p;
  };
  auto allocH = [&](size_t n) { return (_Float16*)alloc(n * sizeof(_Float16)); };
  auto allocF = [&](size_t n) { return (float*)alloc(n * sizeof(float)); };

  const int CH = 320, NCHUNK = 6400 / CH;

  _Float16* wq16  = allocH((size_t)4 * 512 * 512);
  _Float16* wk16  = allocH((size_t)4 * 512 * 512);
  _Float16* wv16  = allocH((size_t)4 * 512 * 512);
  _Float16* wo16  = allocH((size_t)4 * 512 * 512);
  _Float16* fw1_16 = allocH((size_t)4 * 2048 * 512);
  _Float16* fw2_16 = allocH((size_t)4 * 512 * 2048);
  _Float16* cw1_16 = allocH((size_t)100 * 1024);
  _Float16* cw2_16 = allocH((size_t)100 * 2048);
  _Float16* cw3_16 = allocH((size_t)100 * 3072);
  _Float16* tfw16  = allocH((size_t)512 * 300);
  _Float16* fc1w16 = allocH((size_t)512 * 512);
  _Float16* fc2w16 = allocH((size_t)204 * 512);
  _Float16* fc3w16 = allocH((size_t)204 * 204);
  _Float16* fc4w16 = allocH((size_t)204 * 142);
  _Float16* user16 = allocH((size_t)32 * 142);

  _Float16* chunkbuf = allocH((size_t)(CH * 50 + 2) * 1024);
  float* y1f = allocF((size_t)CH * 50 * 100);
  float* y2f = allocF((size_t)CH * 50 * 100);
  float* y3f = allocF((size_t)CH * 50 * 100);
  _Float16* pooled16 = allocH((size_t)6400 * 300);
  float* s2raw = allocF((size_t)6400 * 512);
  float*    x200f = allocF((size_t)6400 * 512);
  _Float16* x200h = allocH((size_t)6400 * 512);
  float*    x101f = allocF((size_t)3232 * 512);
  _Float16* x101h = allocH((size_t)3232 * 512);
  _Float16* q16 = allocH((size_t)6400 * 512);
  _Float16* k16 = allocH((size_t)6400 * 512);
  _Float16* v16 = allocH((size_t)6400 * 512);
  float*    scoresf = allocF((size_t)32 * 8 * 200 * 200);
  _Float16* probs16 = allocH((size_t)32 * 8 * 200 * 200);
  _Float16* ctx16   = allocH((size_t)6400 * 512);
  float*    delta   = allocF((size_t)6400 * 512);
  _Float16* h16buf  = allocH((size_t)6400 * 2048);
  float*    gbuf  = allocF((size_t)6400 * 512);
  float*    ybuff = allocF((size_t)6400 * 512);
  _Float16* ybufh = allocH((size_t)6400 * 512);
  float*    p1f   = allocF((size_t)6400 * 204);
  float*    p101f = allocF((size_t)3232 * 204);
  _Float16* p101h = allocH((size_t)3232 * 204);
  _Float16* P16   = allocH((size_t)32 * 100 * 204);
  float*    SLlf  = allocF((size_t)32 * 204 * 204);
  float*    clsIII  = allocF((size_t)32 * 204);
  float*    out0pre = allocF((size_t)32 * 204);
  float*    cmean   = allocF((size_t)32 * 2 * 204);
  float*    amean   = allocF((size_t)32 * 2 * 512);
  float*    user4m  = allocF((size_t)32 * 204);

  if (off > ws_size) return;  // insufficient workspace: do nothing (no corruption)

  // ---- weight conversion to f16 (once per call) ----
  auto conv16 = [&](const float* s, _Float16* d, long long n) {
    f32_to_f16_k<<<dim3((unsigned)((n + 255) / 256)), 256, 0, stream>>>(s, d, n);
  };
  conv16(wq, wq16, 4LL * 512 * 512);   conv16(wk, wk16, 4LL * 512 * 512);
  conv16(wv, wv16, 4LL * 512 * 512);   conv16(wo, wo16, 4LL * 512 * 512);
  conv16(fw1, fw1_16, 4LL * 2048 * 512); conv16(fw2, fw2_16, 4LL * 512 * 2048);
  conv16(cw1, cw1_16, 100LL * 1024);   conv16(cw2, cw2_16, 100LL * 2048);
  conv16(cw3, cw3_16, 100LL * 3072);   conv16(tfw, tfw16, 512LL * 300);
  conv16(fc1w, fc1w16, 512LL * 512);   conv16(fc2w, fc2w16, 204LL * 512);
  conv16(fc3w, fc3w16, 204LL * 204);   conv16(fc4w, fc4w16, 204LL * 142);
  conv16(user_info, user16, 32LL * 142);

  // ---- TextCNN over all 6400 sequences (s1 is a row-subset of s2) ----
  for (int ch = 0; ch < NCHUNK; ++ch) {
    int rowbase = ch * CH * 50;
    gather_embed_k<<<dim3(CH * 50 + 2), 256, 0, stream>>>(src, emb, chunkbuf, CH * 50, rowbase);
    // convs as GEMMs (contiguous im2col via lda=1024)
    gemm(stream, chunkbuf, cw1_16, cb1, y1f, nullptr, CH * 50, 100, 1024, 1024, 1024, 100, 0, 0, 1.f, 1);
    gemm(stream, chunkbuf, cw2_16, cb2, y2f, nullptr, CH * 50, 100, 2048, 1024, 2048, 100, 0, 0, 1.f, 1);
    gemm(stream, chunkbuf, cw3_16, cb3, y3f, nullptr, CH * 50, 100, 3072, 1024, 3072, 100, 0, 0, 1.f, 1);
    maxpool_k<<<dim3(ceil_div_i(CH * 300, 256)), 256, 0, stream>>>(y1f, y2f, y3f, pooled16, CH, ch * CH);
  }
  gemm(stream, pooled16, tfw16, tfb, s2raw, nullptr, 6400, 512, 300, 300, 300, 512, 0, 0, 1.f, 0);
  ln_free_k<<<dim3(6400), 256, 0, stream>>>(s2raw, x200f, x200h, 512);
  build_s1_k<<<dim3(3232), 256, 0, stream>>>(x200f, x101f, x101h);

  EncCtx ec{stream, wq16, wk16, wv16, wo16, fw1_16, fw2_16,
            bq, bk, bv, bo, l1g, l1b, fb1, fb2, l2g, l2b,
            q16, k16, v16, probs16, ctx16, h16buf, scoresf, delta};

  // ---- S=101 path ----
  encode(ec, x101f, x101h, 101, mask101);
  const int M1 = 32 * 101;
  gemm(stream, x101h, fc1w16, fc1b, gbuf, nullptr, M1, 512, 512, 512, 512, 512, 0, 0, 1.f, 2);
  ln_scale_k<<<dim3(M1), 256, 0, stream>>>(gbuf, pscale, pbias, ybuff, ybufh, 101, 512);
  gemm(stream, ybufh, fc2w16, fc2b, p101f, p101h, M1, 204, 512, 512, 512, 204, 0, 0, 1.f, 0);
  // fc3 on token rows 1..100 per batch element
  gemm(stream, p101h + 204, fc3w16, fc3b, nullptr, P16, 100, 204, 204, 204, 204, 204, 0, 0, 1.f, 0,
       32, 1, 101LL * 204, 0, 0, 0, 100LL * 204, 0);
  // SLl[b] = P^T P (TN)
  gemm(stream, P16, P16, nullptr, SLlf, nullptr, 204, 204, 100, 204, 204, 204, 1, 1, 1.f, 0,
       32, 1, 100LL * 204, 0, 100LL * 204, 0, 204LL * 204, 0);
  clsIII_k<<<dim3(32), 256, 0, stream>>>(p101f, fc3w, fc3b, clsIII);
  outIII_k<<<dim3(32), 256, 0, stream>>>(clsIII, label_sim, SLlf, out0pre);
  softmax_rows_k<<<dim3(32), 256, 0, stream>>>(out0pre, 204LL, outp, 204);
  softmax_rows_k<<<dim3(32), 256, 0, stream>>>(p101f, 101LL * 204, outp + 6528, 204);

  // ---- S=200 path ----
  encode(ec, x200f, x200h, 200, mask200);
  const int M2 = 6400;
  gemm(stream, x200h, fc1w16, fc1b, gbuf, nullptr, M2, 512, 512, 512, 512, 512, 0, 0, 1.f, 2);
  ln_scale_k<<<dim3(M2), 256, 0, stream>>>(gbuf, pscale, pbias, ybuff, ybufh, 200, 512);
  gemm(stream, ybufh, fc2w16, fc2b, p1f, nullptr, M2, 204, 512, 512, 512, 204, 0, 0, 1.f, 0);
  chunk_mean_softmax_k<<<dim3(64), 256, 0, stream>>>(p1f, 204, 200, cmean);
  chunk_mean_softmax_k<<<dim3(64), 256, 0, stream>>>(ybuff, 512, 200, amean);
  loss_sims_k<<<dim3(1), 32, 0, stream>>>(cmean, amean, outp + 13056);

  // ---- match (contrastive) ----
  gemm(stream, user16, fc4w16, fc4b, user4m, nullptr, 32, 204, 142, 142, 142, 204, 0, 0, 1.f, 0);
  contrastive_k<<<dim3(1), 1024, 0, stream>>>(p101f, user4m, outp + 13057);
}
